// MultiHeadAttention_33054068310185
// MI455X (gfx1250) — matrix operand order repaired and checked by reading, not compiled, not run on hardware
//
#include <hip/hip_runtime.h>
#include <hip/hip_bf16.h>

#define DEV __device__ __forceinline__

typedef __attribute__((ext_vector_type(16))) _Float16 v16h;
typedef __attribute__((ext_vector_type(8)))  float    v8f;
typedef __attribute__((ext_vector_type(4))) unsigned  u32x4;
typedef __attribute__((ext_vector_type(8))) unsigned  u32x8;

constexpr int Bv = 2, Lv = 2048, Dv = 1024, Hv = 16, DHv = 64;
constexpr int BL = Bv * Lv;

// ---------------------------------------------------------------------------
// WMMA wrapper: D = A(16x32 f16) x B(32x16 f16) + C(16x16 f32)
// ---------------------------------------------------------------------------
DEV v8f wmma32(v16h a, v16h b, v8f c) {
  return __builtin_amdgcn_wmma_f32_16x16x32_f16(
      /*neg_a=*/false, a, /*neg_b=*/false, b,
      /*c_mod=*/(short)0, c, /*reuse_a=*/false, /*reuse_b=*/false);
}

// A fragment (16x32, 16-bit, wave32): lane m = lane&15; half-wave picks
// K-octets {0..7,16..23} (lanes 0-15) or {8..15,24..31} (lanes 16-31).
DEV v16h frag_a(const _Float16* base, int ld) {
  const int lane = threadIdx.x & 31;
  const _Float16* p = base + (lane & 15) * ld + ((lane >> 4) << 3);
  union { uint4 u[2]; v16h h; } c;
  c.u[0] = *reinterpret_cast<const uint4*>(p);
  c.u[1] = *reinterpret_cast<const uint4*>(p + 16);
  return c.h;
}

// B fragment (32x16, 16-bit, wave32): lane n = lane&15 selects column; lanes
// 0-15 hold K={0..7,16..23}, lanes 16-31 hold K={8..15,24..31}.
DEV v16h frag_b(const _Float16* base, int ld) {
  const int lane = threadIdx.x & 31;
  const _Float16* p = base + (lane & 15) * ld + ((lane >> 4) << 3);
  union { uint4 u[2]; v16h h; } c;
  c.u[0] = *reinterpret_cast<const uint4*>(p);
  c.u[1] = *reinterpret_cast<const uint4*>(p + 16);
  return c.h;
}

// ---------------------------------------------------------------------------
// CDNA5 async memory -> LDS copy (ASYNCcnt-tracked, bypasses VGPRs).
// ---------------------------------------------------------------------------
DEV void async_g2l_b128(void* lds_dst, const void* gsrc) {
  const unsigned loff = (unsigned)(uintptr_t)lds_dst;
  const unsigned long long ga = (unsigned long long)(uintptr_t)gsrc;
  asm volatile("global_load_async_to_lds_b128 %0, %1, off"
               :: "v"(loff), "v"(ga)
               : "memory");
}
DEV void wait_async0() {
  asm volatile("s_wait_asynccnt 0x0" ::: "memory");
}

// ---------------------------------------------------------------------------
// CDNA5 Tensor Data Mover: DMA a 2D tile (tile_d0 elements x tile_d1 rows of
// f16, row stride ld elements) from global into LDS, padding each 64B LDS row
// to 80B (pad_interval=3 -> every 16 DWORDs, pad_amount=3 -> 4 DWORDs) so the
// LDS layout matches LDT=40 halves. Issued once per wave; TENSORcnt-tracked.
// D# layout per CDNA5 ISA 8.3/8.4 (groups 2/3 zero-filled: 2D tensor).
// ---------------------------------------------------------------------------
DEV void tdm_load_tile_f16(void* lds_dst, const void* gsrc,
                           unsigned tile_d0, unsigned tile_d1,
                           unsigned tensor_d0, unsigned tensor_d1,
                           unsigned ld) {
  const unsigned long long ga = (unsigned long long)(uintptr_t)gsrc;
  u32x4 g0;
  g0[0] = 1u;                                        // count=1 (valid), user mode
  g0[1] = (unsigned)(uintptr_t)lds_dst;              // lds_addr (bytes)
  g0[2] = (unsigned)ga;                              // global_addr[31:0]
  g0[3] = ((unsigned)(ga >> 32) & 0x01FFFFFFu)       // global_addr[56:32]
          | (2u << 30);                              // type=2 ("image")
  u32x8 g1;
  g1[0] = (1u << 16)                                 // data_size=1 (2 bytes)
          | (1u << 20)                               // pad_enable
          | (3u << 22)                               // pad_interval: 16 DWORDs
          | (3u << 25);                              // pad_amount: 4 DWORDs
  g1[1] = (tensor_d0 & 0xFFFFu) << 16;               // tensor_dim0[15:0]
  g1[2] = (tensor_d0 >> 16) | ((tensor_d1 & 0xFFFFu) << 16);
  g1[3] = (tensor_d1 >> 16) | (tile_d0 << 16);       // tile_dim0
  g1[4] = tile_d1;                                   // tile_dim1 | tile_dim2=0
  g1[5] = ld;                                        // tensor_dim0_stride[31:0]
  g1[6] = 0u;
  g1[7] = 0u;
  const u32x4 gz = {0u, 0u, 0u, 0u};
  asm volatile("tensor_load_to_lds %0, %1, %2, %3"
               :: "s"(g0), "s"(g1), "s"(gz), "s"(gz)
               : "memory");
}

// ---------------------------------------------------------------------------
// fp32 -> f16 conversion of the four weight matrices
// ---------------------------------------------------------------------------
__global__ void cvt4_kernel(const float* __restrict__ a, const float* __restrict__ b,
                            const float* __restrict__ c, const float* __restrict__ d,
                            _Float16* __restrict__ oa, _Float16* __restrict__ ob,
                            _Float16* __restrict__ oc, _Float16* __restrict__ od) {
  const int i = blockIdx.x * blockDim.x + threadIdx.x;
  if (i < Dv * Dv) {
    oa[i] = (_Float16)a[i];
    ob[i] = (_Float16)b[i];
    oc[i] = (_Float16)c[i];
    od[i] = (_Float16)d[i];
  }
}

// ---------------------------------------------------------------------------
// GEMM: C[m,n] = sum_k A[m,k] * W[n,k] + bias[n]   (M=4096, N=1024, K=1024)
// Block tile 64x64, 128 threads (4 waves), each wave a 32x32 quadrant.
// B tile staged by the Tensor Data Mover (wave 0); A tile staged by per-lane
// async-to-LDS (f16) or manual convert (fp32 source).
// OUT_MODE 0: f16 head-split [B,H,L,DH]   (Q, K projections)
// OUT_MODE 1: f16 head-split transposed [B,H,DH,L]   (V projection)
// OUT_MODE 2: fp32 plain [BL, D]          (output projection)
// ---------------------------------------------------------------------------
template <int OUT_MODE, bool A_F16>
__global__ __launch_bounds__(128) void gemm_kernel(
    const float* __restrict__ Af, const _Float16* __restrict__ Ah,
    const _Float16* __restrict__ W, const float* __restrict__ bias,
    _Float16* __restrict__ outH, float* __restrict__ outF) {
  constexpr int LDT = 40;  // padded LDS row stride in halves (80B, 16B-aligned)
  __shared__ _Float16 As[64 * LDT];
  __shared__ _Float16 Bs[64 * LDT];

  const int n0 = blockIdx.x * 64;
  const int m0 = blockIdx.y * 64;
  const int tid = threadIdx.x;
  const int wid = tid >> 5;
  const int wm = (wid >> 1) * 32, wn = (wid & 1) * 32;
  const int srow = tid >> 1;        // 0..63 staging row
  const int scol = (tid & 1) * 16;  // 0 / 16 staging k-chunk

  v8f acc[4] = {};

  for (int k0 = 0; k0 < Dv; k0 += 32) {
    // --- stage B tile (64 rows x 32 halves of W) via TDM, wave 0 only ---
    if (wid == 0) {
      tdm_load_tile_f16(&Bs[0], W + (size_t)n0 * Dv + k0,
                        /*tile_d0=*/32u, /*tile_d1=*/64u,
                        /*tensor_d0=*/(unsigned)Dv, /*tensor_d1=*/(unsigned)Dv,
                        /*ld=*/(unsigned)Dv);
    }
    // --- stage A tile (64x32) ---
    if (A_F16) {
      const _Float16* ga = Ah + (size_t)(m0 + srow) * Dv + k0 + scol;
      async_g2l_b128(&As[srow * LDT + scol], ga);
      async_g2l_b128(&As[srow * LDT + scol + 8], ga + 8);
    } else {
      const float4* ap = reinterpret_cast<const float4*>(
          Af + (size_t)(m0 + srow) * Dv + k0 + scol);
#pragma unroll
      for (int j = 0; j < 4; ++j) {
        float4 v = ap[j];
        _Float16* d = &As[srow * LDT + scol + j * 4];
        d[0] = (_Float16)v.x; d[1] = (_Float16)v.y;
        d[2] = (_Float16)v.z; d[3] = (_Float16)v.w;
      }
    }
    // prefetch next K tile into cache (global_prefetch_b8)
    if (k0 + 32 < Dv) {
      __builtin_prefetch(W + (size_t)(n0 + srow) * Dv + k0 + 32 + scol, 0, 0);
      if (A_F16)
        __builtin_prefetch(Ah + (size_t)(m0 + srow) * Dv + k0 + 32 + scol, 0, 0);
      else
        __builtin_prefetch(Af + (size_t)(m0 + srow) * Dv + k0 + 32 + scol, 0, 0);
    }
    if (A_F16) wait_async0();
    if (wid == 0) __builtin_amdgcn_s_wait_tensorcnt(0);
    __syncthreads();

    const v16h a0 = frag_a(&As[(wm + 0) * LDT], LDT);
    const v16h a1 = frag_a(&As[(wm + 16) * LDT], LDT);
    const v16h b0 = frag_b(&Bs[(wn + 0) * LDT], LDT);
    const v16h b1 = frag_b(&Bs[(wn + 16) * LDT], LDT);
    acc[0] = wmma32(a0, b0, acc[0]);
    acc[1] = wmma32(a0, b1, acc[1]);
    acc[2] = wmma32(a1, b0, acc[2]);
    acc[3] = wmma32(a1, b1, acc[3]);
    __syncthreads();
  }

  // epilogue: C/D layout -> lane = N (lane&15), VGPR r -> M = 8*(lane>=16)+r
  const int lane = tid & 31;
#pragma unroll
  for (int t = 0; t < 4; ++t) {
    const int row0 = m0 + wm + (t >> 1) * 16;
    const int ncol = n0 + wn + (t & 1) * 16 + (lane & 15);
    const float bv = bias[ncol];
#pragma unroll
    for (int r = 0; r < 8; ++r) {
      const int mrow = row0 + ((lane >> 4) << 3) + r;
      const float val = acc[t][r] + bv;
      const int bb = mrow >> 11, lseq = mrow & (Lv - 1);
      const int hh = ncol >> 6, dh = ncol & (DHv - 1);
      if (OUT_MODE == 0) {
        outH[((size_t)(bb * Hv + hh) * Lv + lseq) * DHv + dh] = (_Float16)val;
      } else if (OUT_MODE == 1) {
        outH[((size_t)(bb * Hv + hh) * DHv + dh) * Lv + lseq] = (_Float16)val;
      } else {
        outF[(size_t)mrow * Dv + ncol] = val;
      }
    }
  }
}

// ---------------------------------------------------------------------------
// Flash attention: block = 64 queries of one (b,h); 4 waves x 16 queries.
// Streams 32-key blocks: S = Q K^T (4 wmma), scale*bias + mask, online
// softmax through per-wave LDS tile, O += P V (4 wmma, V pre-transposed).
// ---------------------------------------------------------------------------
__global__ __launch_bounds__(128) void attn_kernel(
    const _Float16* __restrict__ Qh,  // [B,H,L,DH]
    const _Float16* __restrict__ Kh,  // [B,H,L,DH]
    const _Float16* __restrict__ Vt,  // [B,H,DH,L]
    const float* __restrict__ bias,   // [B,L,L]
    const int* __restrict__ mask,     // [B,L]
    _Float16* __restrict__ out) {     // [B,L,D] = [b,l,h*DH+dh], f16
  constexpr int SLD = 36;  // padded fp32 row stride (144B, 16B-aligned)
  __shared__ float sS[4][16 * SLD];
  __shared__ float sAlpha[4][16];
  __shared__ float sLsum[4][16];

  const int qt = blockIdx.x, h = blockIdx.y, b = blockIdx.z;
  const int tid = threadIdx.x, w = tid >> 5, lane = tid & 31;
  const int hi = lane >> 4;          // half-wave select
  const int nl = lane & 15;          // column / row-owner index
  const int q0 = qt * 64 + w * 16;
  const size_t bh = (size_t)(b * Hv + h);

  const _Float16* Qbase = Qh + (bh * Lv + q0) * DHv;
  const _Float16* Kbase = Kh + bh * Lv * DHv;
  const _Float16* Vbase = Vt + bh * (size_t)DHv * Lv;
  float* sw = &sS[w][0];

  const v16h qa0 = frag_a(Qbase, DHv);       // Q, K-dim chunk 0..31
  const v16h qa1 = frag_a(Qbase + 32, DHv);  // Q, K-dim chunk 32..63

  v8f o[4] = {};
  float m_run = -3.0e38f, l_run = 0.0f;

  for (int kb = 0; kb < Lv; kb += 32) {
    const _Float16* kp = Kbase + (size_t)kb * DHv;
    if (kb + 32 < Lv)  // prefetch next 32 K rows (one 128B line per lane)
      __builtin_prefetch(Kbase + (size_t)(kb + 32 + lane) * DHv, 0, 0);

    // ---- S = Q K^T  (two 16-key column tiles, DH=64 as 2 k-steps) ----
    v8f s0 = {}, s1 = {};
    s0 = wmma32(qa0, frag_b(kp, DHv), s0);
    s0 = wmma32(qa1, frag_b(kp + 32, DHv), s0);
    s1 = wmma32(qa0, frag_b(kp + 16 * DHv, DHv), s1);
    s1 = wmma32(qa1, frag_b(kp + 16 * DHv + 32, DHv), s1);

    // ---- scale * bias, key mask, spill to per-wave LDS tile ----
    {
      const int key0 = kb + nl, key1 = key0 + 16;
      const int mk0 = mask[b * Lv + key0];
      const int mk1 = mask[b * Lv + key1];
#pragma unroll
      for (int r = 0; r < 8; ++r) {
        const int mrow = hi * 8 + r;
        const int qg = q0 + mrow;
        const size_t brow = ((size_t)b * Lv + qg) * Lv;
        float v0 = s0[r] * 0.125f * bias[brow + key0];
        float v1 = s1[r] * 0.125f * bias[brow + key1];
        if (mk0 == 0) v0 = -10000.0f;
        if (mk1 == 0) v1 = -10000.0f;
        sw[mrow * SLD + nl] = v0;
        sw[mrow * SLD + nl + 16] = v1;
      }
    }
    __syncthreads();

    // ---- online softmax stats for row nl (duplicated per half-wave) ----
    {
      const float4* rowp = reinterpret_cast<const float4*>(&sw[nl * SLD]);
      float4 rv[8];
#pragma unroll
      for (int j = 0; j < 8; ++j) rv[j] = rowp[j];
      float tmax = -3.0e38f;
#pragma unroll
      for (int j = 0; j < 8; ++j)
        tmax = fmaxf(tmax, fmaxf(fmaxf(rv[j].x, rv[j].y), fmaxf(rv[j].z, rv[j].w)));
      const float m_new = fmaxf(m_run, tmax);
      const float alpha = __expf(m_run - m_new);
      float psum = 0.0f;
#pragma unroll
      for (int j = 0; j < 8; ++j)
        psum += __expf(rv[j].x - m_new) + __expf(rv[j].y - m_new) +
                __expf(rv[j].z - m_new) + __expf(rv[j].w - m_new);
      l_run = l_run * alpha + psum;
      m_run = m_new;
      sAlpha[w][nl] = alpha;
    }
    __syncthreads();

    // ---- P fragment (A layout, row nl == this lane's stats row) ----
    v16h p;
    {
      const int klo = hi * 8;
#pragma unroll
      for (int i = 0; i < 8; ++i)
        p[i] = (_Float16)__expf(sw[nl * SLD + klo + i] - m_run);
#pragma unroll
      for (int i = 0; i < 8; ++i)
        p[8 + i] = (_Float16)__expf(sw[nl * SLD + 16 + klo + i] - m_run);
    }

    // ---- rescale O accumulators by alpha(row) ----
#pragma unroll
    for (int r = 0; r < 8; ++r) {
      const float av = sAlpha[w][hi * 8 + r];
#pragma unroll
      for (int t = 0; t < 4; ++t) o[t][r] *= av;
    }

    // ---- O += P @ V  (V transposed -> contiguous b128 B-fragments) ----
#pragma unroll
    for (int t = 0; t < 4; ++t) {
      const v16h vb = frag_b(Vbase + (size_t)(t * 16) * Lv + kb, Lv);
      o[t] = wmma32(p, vb, o[t]);
    }
    __syncthreads();
  }

  sLsum[w][nl] = l_run;
  __syncthreads();

  // ---- normalize and store f16 [b, l, h*DH+dh] ----
#pragma unroll
  for (int t = 0; t < 4; ++t) {
    const int dh = t * 16 + nl;
#pragma unroll
    for (int r = 0; r < 8; ++r) {
      const int row = hi * 8 + r;
      const float val = o[t][r] / sLsum[w][row];
      out[((size_t)(b * Lv) + q0 + row) * Dv + h * DHv + dh] = (_Float16)val;
    }
  }
}

// ---------------------------------------------------------------------------
// Host launcher
// ---------------------------------------------------------------------------
extern "C" void kernel_launch(void* const* d_in, const int* in_sizes, int n_in,
                              void* d_out, int out_size, void* d_ws, size_t ws_size,
                              hipStream_t stream) {
  const float* q    = (const float*)d_in[0];
  const float* k    = (const float*)d_in[1];
  const float* v    = (const float*)d_in[2];
  const float* bias = (const float*)d_in[3];
  const int*   mask = (const int*)d_in[4];
  const float* Wq = (const float*)d_in[5];
  const float* bq = (const float*)d_in[6];
  const float* Wk = (const float*)d_in[7];
  const float* bk = (const float*)d_in[8];
  const float* Wv = (const float*)d_in[9];
  const float* bv = (const float*)d_in[10];
  const float* Wo = (const float*)d_in[11];
  const float* bo = (const float*)d_in[12];
  float* out = (float*)d_out;

  // ws layout (halves): 4x 1M weights, then Qh/Kh/Vt/AO at 4M each = 40MB
  _Float16* ws = (_Float16*)d_ws;
  const size_t M1 = (size_t)1 << 20;
  _Float16* wq16 = ws + 0 * M1;
  _Float16* wk16 = ws + 1 * M1;
  _Float16* wv16 = ws + 2 * M1;
  _Float16* wo16 = ws + 3 * M1;
  _Float16* Qh = ws + 4 * M1;
  _Float16* Kh = ws + 8 * M1;
  _Float16* Vt = ws + 12 * M1;
  _Float16* AO = ws + 16 * M1;

  cvt4_kernel<<<dim3(Dv * Dv / 256), 256, 0, stream>>>(Wq, Wk, Wv, Wo,
                                                       wq16, wk16, wv16, wo16);

  const dim3 gg(Dv / 64, BL / 64);
  gemm_kernel<0, false><<<gg, 128, 0, stream>>>(q, nullptr, wq16, bq, Qh, nullptr);
  gemm_kernel<0, false><<<gg, 128, 0, stream>>>(k, nullptr, wk16, bk, Kh, nullptr);
  gemm_kernel<1, false><<<gg, 128, 0, stream>>>(v, nullptr, wv16, bv, Vt, nullptr);

  attn_kernel<<<dim3(Lv / 64, Hv, Bv), 128, 0, stream>>>(Qh, Kh, Vt, bias, mask, AO);

  gemm_kernel<2, true><<<gg, 128, 0, stream>>>(nullptr, AO, wo16, bo, nullptr, out);
}
